// BatchAwareCrossAttention_28956669510264
// MI455X (gfx1250) — compile-verified, matrix operand order checked by reading, not run on hardware
//
#include <hip/hip_runtime.h>
#include <hip/hip_bf16.h>
#include <math.h>

#define B_  4096
#define N_  64
#define D_  256
#define H_  8
#define NB_ 16
#define DH_ 32

typedef __bf16 bf16_t;
typedef __attribute__((ext_vector_type(16))) __bf16 v16bf;
typedef __attribute__((ext_vector_type(8)))  __bf16 v8bf;
typedef __attribute__((ext_vector_type(8)))  float  v8f;

__device__ __forceinline__ v8f v8f_zero() {
    v8f v;
#pragma unroll
    for (int i = 0; i < 8; ++i) v[i] = 0.0f;
    return v;
}

// A fragment (16x32 bf16, row-major source with row stride D_):
// lanes 0-15 : row = mr+lane,    K chunks [k0, k0+8) and [k0+16, k0+24)
// lanes 16-31: row = mr+lane-16, K chunks [k0+8, k0+16) and [k0+24, k0+32)
__device__ __forceinline__ v16bf frag_a(const bf16_t* __restrict__ A, int mr, int k0, int lane) {
    int r  = mr + (lane & 15);
    int kk = k0 + ((lane >> 4) << 3);
    v8bf lo = *(const v8bf*)(A + r * D_ + kk);
    v8bf hi = *(const v8bf*)(A + r * D_ + kk + 16);
    v16bf f;
#pragma unroll
    for (int i = 0; i < 8; ++i) { f[i] = lo[i]; f[i + 8] = hi[i]; }
    return f;
}

// B fragment (32x16 bf16). B[k][n] = W[n][k] (we compute X @ W.T), so a
// column of B is a contiguous row of W (row-major [D_][D_] bf16).
__device__ __forceinline__ v16bf frag_b(const bf16_t* __restrict__ W, int nc, int k0, int lane) {
    int c  = nc + (lane & 15);
    int kk = k0 + ((lane >> 4) << 3);
    v8bf lo = *(const v8bf*)(W + c * D_ + kk);
    v8bf hi = *(const v8bf*)(W + c * D_ + kk + 16);
    v16bf f;
#pragma unroll
    for (int i = 0; i < 8; ++i) { f[i] = lo[i]; f[i + 8] = hi[i]; }
    return f;
}

// ---------------- prep 1: convert Wq/Wk/Wv to bf16 (contiguous dst) --------
extern "C" __global__ void __launch_bounds__(256)
baca_cvt_w(const float* __restrict__ Wq, const float* __restrict__ Wk,
           const float* __restrict__ Wv, bf16_t* __restrict__ dst) {
    int idx = blockIdx.x * 256 + threadIdx.x;           // 0 .. 3*65536
    const float* src = (idx < 65536) ? Wq : (idx < 131072 ? Wk : Wv);
    dst[idx] = (bf16_t)src[idx & 65535];
}

// ---------------- prep 2: P_u* = batch_emb @ U*.T + b* + bu* ---------------
extern "C" __global__ void __launch_bounds__(256)
baca_pu(const float* __restrict__ Uq, const float* __restrict__ buq, const float* __restrict__ bq,
        const float* __restrict__ Uk, const float* __restrict__ buk, const float* __restrict__ bk,
        const float* __restrict__ Uv, const float* __restrict__ buv, const float* __restrict__ bv,
        const float* __restrict__ batch_emb,
        float* __restrict__ P_uq, float* __restrict__ P_uk, float* __restrict__ P_uv) {
    int idx = blockIdx.x * 256 + threadIdx.x;           // 0 .. 3*16*256
    int which = idx / (NB_ * D_);
    int rem   = idx - which * (NB_ * D_);
    int nb = rem >> 8, d = rem & 255;
    const float* U  = (which == 0) ? Uq  : (which == 1) ? Uk  : Uv;
    const float* bu = (which == 0) ? buq : (which == 1) ? buk : buv;
    const float* bw = (which == 0) ? bq  : (which == 1) ? bk  : bv;
    float*       P  = (which == 0) ? P_uq: (which == 1) ? P_uk: P_uv;
    float acc = bu[d] + bw[d];
    const float* e = batch_emb + nb * D_;
    const float* u = U + d * D_;
#pragma unroll 8
    for (int k = 0; k < D_; ++k) acc = fmaf(e[k], u[k], acc);
    P[rem] = acc;
}

// ---------------- kernel A: Q = h_recv@Wq.T + gather(P_uq); b1 MLP ---------
// LDS carve: Hs f32[64*256] | h1 f32[64*128] | As bf16[64*256] | br int[64]
extern "C" __global__ void __launch_bounds__(256)
baca_qb1(const float* __restrict__ h_recv, const int* __restrict__ batch_recv,
         const bf16_t* __restrict__ Wq_bf, const float* __restrict__ P_uq,
         const float* __restrict__ batch_emb,
         const float* __restrict__ D1, const float* __restrict__ db1,
         const float* __restrict__ D2, const float* __restrict__ db2,
         float* __restrict__ Qf, float* __restrict__ b1w) {
    extern __shared__ char smem[];
    float*  Hs = (float*)smem;                   // 64*256
    float*  h1 = Hs + 64 * 256;                  // 64*128
    bf16_t* As = (bf16_t*)(h1 + 64 * 128);       // 64*256
    int*    br = (int*)(As + 64 * 256);          // 64

    const int t = threadIdx.x, lane = t & 31, wave = t >> 5;
    const int m0 = blockIdx.x * 64;

    const float* hr = h_recv + (size_t)m0 * D_;
#pragma unroll 4
    for (int i = 0; i < 64; ++i) {
        int idx = t + i * 256;
        float v = hr[idx];
        Hs[idx] = v;
        As[idx] = (bf16_t)v;
    }
    if (t < 64) br[t] = batch_recv[m0 + t];
    __syncthreads();

    // receiver_state = h_recv + E_recv (in place; WMMA below only reads As)
#pragma unroll 4
    for (int i = 0; i < 64; ++i) {
        int idx = t + i * 256;
        Hs[idx] += batch_emb[br[idx >> 8] * D_ + (idx & 255)];
    }

    // WMMA: 64x256 output, 8 waves x 8 tiles
    {
        const int strip = (wave & 3) * 16;
        const int chalf = (wave >> 2) * 8;
        v8f acc[8];
#pragma unroll
        for (int ct = 0; ct < 8; ++ct) acc[ct] = v8f_zero();
        for (int k0 = 0; k0 < D_; k0 += 32) {
            v16bf a = frag_a(As, strip, k0, lane);
#pragma unroll
            for (int ct = 0; ct < 8; ++ct) {
                v16bf bb = frag_b(Wq_bf, (chalf + ct) * 16, k0, lane);
                acc[ct] = __builtin_amdgcn_wmma_f32_16x16x32_bf16(
                    false, a, false, bb, (short)0, acc[ct], false, false);
            }
        }
        const int rbase = strip + ((lane >> 4) << 3);
#pragma unroll
        for (int ct = 0; ct < 8; ++ct) {
            int col = (chalf + ct) * 16 + (lane & 15);
#pragma unroll
            for (int j = 0; j < 8; ++j) {
                int row = rbase + j;
                Qf[(size_t)(m0 + row) * D_ + col] = acc[ct][j] + P_uq[br[row] * D_ + col];
            }
        }
    }
    __syncthreads();

    // hidden = relu(rs @ D1.T + db1)   (8192 outputs, j uniform per wave-pair)
    for (int rep = 0; rep < 32; ++rep) {
        int idx = rep * 256 + t;
        int j = idx >> 6, r = idx & 63;
        float d = db1[j];
        const float* w = D1 + j * D_;
        const float* x = Hs + r * D_;
#pragma unroll 8
        for (int k = 0; k < D_; ++k) d = fmaf(x[k], w[k], d);
        h1[r * 128 + j] = fmaxf(d, 0.0f);
    }
    __syncthreads();

    // b1 = softplus(h1 @ D2.T + db2)
#pragma unroll
    for (int rep = 0; rep < 2; ++rep) {
        int idx = rep * 256 + t;
        int r = idx >> 3, h = idx & 7;
        float d = db2[h];
        const float* w = D2 + h * 128;
        const float* x = h1 + r * 128;
#pragma unroll 8
        for (int k = 0; k < 128; ++k) d = fmaf(x[k], w[k], d);
        b1w[(size_t)(m0 + r) * H_ + h] = log1pf(__expf(d));
    }
}

// ---------------- kernel B: fused K/V projection + attention per b ---------
// V never touches LDS: V-waves keep accumulators in registers and contract
// them against attn directly (register tile j-axis == n reduction axis).
// LDS carve (143 KB -> 2 WGs/WGP):
//   Ks f32[64*256] | Pk f32[16*256] | Pv f32[16*256] | Qs f32[256]
//   sc f32[512] | b1s f32[8] | inv f32[8] | dists f32[64] | zpart f32[8*256]
//   As bf16[64*256] | bs int[64] | msk int[64]
extern "C" __global__ void __launch_bounds__(256)
baca_attn(const float* __restrict__ h_send, const int* __restrict__ batch_recv,
          const int* __restrict__ batch_send, const float* __restrict__ dist,
          const int* __restrict__ mask, const float* __restrict__ gamma,
          const bf16_t* __restrict__ Wk_bf, const bf16_t* __restrict__ Wv_bf,
          const float* __restrict__ P_uk, const float* __restrict__ P_uv,
          const float* __restrict__ Qf, const float* __restrict__ b1w,
          float* __restrict__ z_out, float* __restrict__ attn_out) {
    extern __shared__ char smem[];
    float*  Ks    = (float*)smem;                // 64*256
    float*  Pk    = Ks + 64 * 256;               // 16*256
    float*  Pv    = Pk + 16 * 256;               // 16*256
    float*  Qs    = Pv + 16 * 256;               // 256
    float*  sc    = Qs + 256;                    // 64*8
    float*  b1s   = sc + 512;                    // 8
    float*  inv   = b1s + 8;                     // 8
    float*  dists = inv + 8;                     // 64
    float*  zpart = dists + 64;                  // 8*256
    bf16_t* As    = (bf16_t*)(zpart + 8 * 256);  // 64*256 (16B aligned)
    int*    bs    = (int*)(As + 64 * 256);       // 64
    int*    msk   = bs + 64;                     // 64

    const int t = threadIdx.x, lane = t & 31, wave = t >> 5;
    const int b = blockIdx.x;

    // stage: h_send[b] -> bf16, P_uk/P_uv tables, Q row, per-n scalars
    const float* hs = h_send + (size_t)b * (N_ * D_);
#pragma unroll 4
    for (int i = 0; i < 64; ++i) {
        int idx = t + i * 256;
        As[idx] = (bf16_t)hs[idx];
    }
    for (int i = t; i < NB_ * D_; i += 256) { Pk[i] = P_uk[i]; Pv[i] = P_uv[i]; }
    Qs[t] = Qf[(size_t)b * D_ + t];
    if (t < 64) {
        bs[t]    = batch_send[b * N_ + t];
        msk[t]   = mask[b * N_ + t];
        dists[t] = dist[b * N_ + t];
    }
    if (t < H_) b1s[t] = b1w[(size_t)b * H_ + t];
    __syncthreads();

    // K/V projection: waves 0-3 -> K strips 0..3 (to LDS with P_uk bias),
    //                 waves 4-7 -> V strips 0..3 (stay in registers)
    const int mat   = wave >> 2;
    const int strip = (wave & 3) * 16;
    const int rbase = strip + ((lane >> 4) << 3);
    v8f acc[16];
#pragma unroll
    for (int ct = 0; ct < 16; ++ct) acc[ct] = v8f_zero();
    {
        const bf16_t* W = mat ? Wv_bf : Wk_bf;
        for (int k0 = 0; k0 < D_; k0 += 32) {
            v16bf a = frag_a(As, strip, k0, lane);
#pragma unroll
            for (int ct = 0; ct < 16; ++ct) {
                v16bf bb = frag_b(W, ct * 16, k0, lane);
                acc[ct] = __builtin_amdgcn_wmma_f32_16x16x32_bf16(
                    false, a, false, bb, (short)0, acc[ct], false, false);
            }
        }
    }
    if (mat == 0) {
#pragma unroll
        for (int ct = 0; ct < 16; ++ct) {
            int col = ct * 16 + (lane & 15);
#pragma unroll
            for (int j = 0; j < 8; ++j) {
                int row = rbase + j;                 // n index 0..63
                Ks[row * D_ + col] = acc[ct][j] + Pk[bs[row] * D_ + col];
            }
        }
    }
    __syncthreads();

    // scores[n][h]
    const int   br    = batch_recv[b];
    const float scale = 0.1767766952966369f;   // 1/sqrt(32)
#pragma unroll
    for (int rep = 0; rep < 2; ++rep) {
        int idx = t + rep * 256;
        int n = idx >> 3, h = idx & 7;
        float s;
        if (msk[n] == 0) {
            s = -INFINITY;
        } else {
            float d = 0.0f;
            const float* q = Qs + h * DH_;
            const float* k = Ks + n * D_ + h * DH_;
#pragma unroll
            for (int kk = 0; kk < DH_; ++kk) d = fmaf(q[kk], k[kk], d);
            s = d * scale - b1s[h] * dists[n] + gamma[br * NB_ + bs[n]];
        }
        sc[n * H_ + h] = s;
    }
    __syncthreads();

    // softmax over n, per h (8 lanes, N=64 serial — tiny)
    if (t < H_) {
        float m = -INFINITY;
        for (int n = 0; n < N_; ++n) m = fmaxf(m, sc[n * H_ + t]);
        float s = 0.0f;
        for (int n = 0; n < N_; ++n) {
            float e = __expf(sc[n * H_ + t] - m);
            sc[n * H_ + t] = e;
            s += e;
        }
        inv[t] = 1.0f / s;
    }
    __syncthreads();

#pragma unroll
    for (int rep = 0; rep < 2; ++rep) {
        int idx = t + rep * 256;
        int n = idx >> 3, h = idx & 7;
        float a = sc[n * H_ + h] * inv[h];
        sc[n * H_ + h] = a;
        attn_out[((size_t)b * N_ + n) * H_ + h] = a;
    }
    __syncthreads();

    // V-waves: contract register V tiles with attn.
    // Lane owns col = ct*16 + (lane&15); acc[ct][j] is V[rbase+j][col].
    if (mat == 1) {
        const int half = (wave - 4) * 2 + (lane >> 4);   // 0..7
#pragma unroll
        for (int ct = 0; ct < 16; ++ct) {
            int col = ct * 16 + (lane & 15);
            int h = col >> 5;
            float p = 0.0f;
#pragma unroll
            for (int j = 0; j < 8; ++j)
                p = fmaf(sc[(rbase + j) * H_ + h], acc[ct][j], p);
            zpart[half * D_ + col] = p;
        }
    }
    __syncthreads();

    // z[b, col] = sum(zpart) + sum_n attn[n,h] * P_uv[bs[n]][col]
    {
        const int h = t >> 5;
        float z = 0.0f;
#pragma unroll
        for (int p = 0; p < 8; ++p) z += zpart[p * D_ + t];
#pragma unroll 8
        for (int n = 0; n < N_; ++n)
            z = fmaf(sc[n * H_ + h], Pv[bs[n] * D_ + t], z);
        z_out[(size_t)b * D_ + t] = z;
    }
}

// --------------------------------------------------------------------------
extern "C" void kernel_launch(void* const* d_in, const int* in_sizes, int n_in,
                              void* d_out, int out_size, void* d_ws, size_t ws_size,
                              hipStream_t stream) {
    const float* h_recv     = (const float*)d_in[0];
    const float* h_send     = (const float*)d_in[1];
    const int*   batch_recv = (const int*)d_in[2];
    const int*   batch_send = (const int*)d_in[3];
    const float* dist       = (const float*)d_in[4];
    const int*   mask       = (const int*)d_in[5];
    const float* Wq = (const float*)d_in[6];  const float* bq  = (const float*)d_in[7];
    const float* Wk = (const float*)d_in[8];  const float* bk  = (const float*)d_in[9];
    const float* Wv = (const float*)d_in[10]; const float* bv  = (const float*)d_in[11];
    const float* Uq = (const float*)d_in[12]; const float* buq = (const float*)d_in[13];
    const float* Uk = (const float*)d_in[14]; const float* buk = (const float*)d_in[15];
    const float* Uv = (const float*)d_in[16]; const float* buv = (const float*)d_in[17];
    const float* batch_emb = (const float*)d_in[18];
    const float* D1  = (const float*)d_in[19]; const float* db1 = (const float*)d_in[20];
    const float* D2  = (const float*)d_in[21]; const float* db2 = (const float*)d_in[22];
    const float* gamma = (const float*)d_in[23];

    // workspace carve (≈4.6 MB)
    bf16_t* Wq_bf = (bf16_t*)d_ws;
    bf16_t* Wk_bf = Wq_bf + D_ * D_;
    bf16_t* Wv_bf = Wk_bf + D_ * D_;
    float*  P_uq  = (float*)(Wv_bf + D_ * D_);
    float*  P_uk  = P_uq + NB_ * D_;
    float*  P_uv  = P_uk + NB_ * D_;
    float*  Qf    = P_uv + NB_ * D_;            // B*D
    float*  b1w   = Qf + (size_t)B_ * D_;       // B*H

    float* z_out    = (float*)d_out;
    float* attn_out = z_out + (size_t)B_ * D_;

    hipLaunchKernelGGL(baca_cvt_w, dim3(3 * D_ * D_ / 256), dim3(256), 0, stream,
                       Wq, Wk, Wv, Wq_bf);
    hipLaunchKernelGGL(baca_pu, dim3(3 * NB_ * D_ / 256), dim3(256), 0, stream,
                       Uq, buq, bq, Uk, buk, bk, Uv, buv, bv, batch_emb,
                       P_uq, P_uk, P_uv);

    size_t lds_a = (64 * 256 + 64 * 128) * sizeof(float)
                 + 64 * 256 * sizeof(bf16_t) + 64 * sizeof(int);
    hipLaunchKernelGGL(baca_qb1, dim3(B_ / 64), dim3(256), lds_a, stream,
                       h_recv, batch_recv, Wq_bf, P_uq, batch_emb,
                       D1, db1, D2, db2, Qf, b1w);

    size_t lds_b = (64 * 256 + 2 * 16 * 256 + 256 + 512 + 8 + 8 + 64 + 8 * 256) * sizeof(float)
                 + 64 * 256 * sizeof(bf16_t) + 128 * sizeof(int);
    hipLaunchKernelGGL(baca_attn, dim3(B_), dim3(256), lds_b, stream,
                       h_send, batch_recv, batch_send, dist, mask, gamma,
                       Wk_bf, Wv_bf, P_uk, P_uv, Qf, b1w, z_out, attn_out);
}
